// model_1_35648228557278
// MI455X (gfx1250) — matrix operand order repaired and checked by reading, not compiled, not run on hardware
//
#include <hip/hip_runtime.h>

#define Ecnt 8
#define Bsz  4096
#define Dk   128
#define Nn   8192
#define NT   512          // Nn/16 column tiles per expert
#define TILES_PER_E 256   // ceil(Bsz/16)
#define NTHREADS 512      // 16 waves
#define NWAVES   16

typedef __attribute__((ext_vector_type(16))) __bf16        v16bf;
typedef __attribute__((ext_vector_type(8)))  float         v8f;
typedef __attribute__((ext_vector_type(8)))  unsigned int  v8u;

__device__ __forceinline__ __bf16 f32_to_bf16_rz(float f) {
    unsigned u = __builtin_bit_cast(unsigned, f);
    unsigned short h = (unsigned short)(u >> 16);
    return __builtin_bit_cast(__bf16, h);
}
__device__ __forceinline__ float bf16_to_f32(__bf16 h) {
    unsigned u = ((unsigned)__builtin_bit_cast(unsigned short, h)) << 16;
    return __builtin_bit_cast(float, u);
}
__device__ __forceinline__ void split_bf16(float f, __bf16& hi, __bf16& lo) {
    hi = f32_to_bf16_rz(f);
    float r = f - bf16_to_f32(hi);
    lo = f32_to_bf16_rz(r);
}
__device__ __forceinline__ void split_pack(float w0, float w1, unsigned& hw, unsigned& lw) {
    unsigned u0 = __builtin_bit_cast(unsigned, w0);
    unsigned u1 = __builtin_bit_cast(unsigned, w1);
    float r0 = w0 - __builtin_bit_cast(float, u0 & 0xffff0000u);
    float r1 = w1 - __builtin_bit_cast(float, u1 & 0xffff0000u);
    unsigned v0 = __builtin_bit_cast(unsigned, r0);
    unsigned v1 = __builtin_bit_cast(unsigned, r1);
    hw = (u0 >> 16) | (u1 & 0xffff0000u);
    lw = (v0 >> 16) | (v1 & 0xffff0000u);
}

// ---------------- routing ----------------
__global__ void init_counts_k(int* counts) {
    if (threadIdx.x < Ecnt) counts[threadIdx.x] = 0;
}

__global__ void route_k(const float* __restrict__ x,
                        const float* __restrict__ tlo,
                        const float* __restrict__ thi,
                        int* __restrict__ counts,
                        int* __restrict__ perm,
                        float* __restrict__ validf) {
    int b = blockIdx.x * blockDim.x + threadIdx.x;
    if (b >= Bsz) return;
    float t = x[(size_t)b * Dk];
    int e = 0;
    bool valid = false;
    #pragma unroll
    for (int i = 0; i < Ecnt; ++i) {
        bool c = (t > tlo[i]) && (t <= thi[i]);
        if (c && !valid) { e = i; valid = true; }   // argmax of bool = first true
    }
    validf[b] = valid ? 1.0f : 0.0f;
    int pos = atomicAdd(&counts[e], 1);
    perm[e * Bsz + pos] = b;
}

// ---------------- W pre-split/pack into WMMA B-fragment order ----------------
// Layout (v8u units, 32B per lane): index = ((blk*4 + c)*2 + h)*32 + lane
//   blk = e*NT + nt, c = K-chunk (0..3), h = 0:hi 1:lo
__global__ __launch_bounds__(128)
void pack_w_k(const float* __restrict__ W, v8u* __restrict__ wp) {
    const int blk  = blockIdx.x;        // e*NT + nt
    const int e    = blk >> 9;          // /NT
    const int nt   = blk & (NT - 1);
    const int c    = threadIdx.x >> 5;  // wave = K-chunk
    const int lane = threadIdx.x & 31;
    const int half = lane >> 4;
    const int n    = nt * 16 + (lane & 15);
    const float* We = W + (size_t)e * Dk * Nn;

    unsigned hw[8], lw[8];
    #pragma unroll
    for (int v = 0; v < 8; ++v) {
        int k = c * 32 + ((v < 4) ? (2 * v) : (16 + 2 * (v - 4))) + 8 * half;
        float w0 = We[(size_t)k * Nn + n];
        float w1 = We[(size_t)(k + 1) * Nn + n];
        split_pack(w0, w1, hw[v], lw[v]);
    }
    v8u H = {hw[0], hw[1], hw[2], hw[3], hw[4], hw[5], hw[6], hw[7]};
    v8u L = {lw[0], lw[1], lw[2], lw[3], lw[4], lw[5], lw[6], lw[7]};
    size_t base = ((size_t)(blk * 4 + c)) * 2 * 32;
    wp[base + lane]      = H;
    wp[base + 32 + lane] = L;
}

// ---------------- grouped GEMM + softmax ----------------
__global__ __launch_bounds__(NTHREADS)
void moe_k(const float* __restrict__ x,
           const v8u*   __restrict__ wp,
           const float* __restrict__ bias,
           const int*   __restrict__ counts,
           const int*   __restrict__ perm,
           const float* __restrict__ validf,
           float* __restrict__ dummy,      // Nn floats of scratch for sentinel rows
           float* __restrict__ out) {
    const int e = blockIdx.y / TILES_PER_E;
    const int t = blockIdx.y % TILES_PER_E;
    const int cnt = counts[e];
    if (t * 16 >= cnt) return;

    __shared__ int   s_rows[16];
    __shared__ float s_pmax[NWAVES][16];
    __shared__ float s_psum[16][33];
    __shared__ float s_rowmax[16];
    __shared__ float s_rowinv[16];

    const int tid  = threadIdx.x;
    const int wave = tid >> 5;          // 0..15
    const int lane = tid & 31;
    const int m    = lane & 15;
    const int half = lane >> 4;

    if (tid < 16) {
        int j = t * 16 + tid;
        s_rows[tid] = (j < cnt) ? perm[e * Bsz + j] : -1;
    }
    __syncthreads();

    // ---- A fragments: 16x128 x-tile, bf16 hi/lo, 4 K-chunks (one-time) ----
    // 16-bit A 16x32: lane(half,m); VGPR v<4 -> K=2v+8*half, v>=4 -> K=16+2(v-4)+8*half
    const int arow = s_rows[m];
    const float* xrow = x + (size_t)(arow >= 0 ? arow : 0) * Dk;
    v16bf Ahi[4], Alo[4];
    #pragma unroll
    for (int c = 0; c < 4; ++c) {
        #pragma unroll
        for (int v = 0; v < 8; ++v) {
            int k = c * 32 + ((v < 4) ? (2 * v) : (16 + 2 * (v - 4))) + 8 * half;
            float f0 = (arow >= 0) ? xrow[k]     : 0.0f;
            float f1 = (arow >= 0) ? xrow[k + 1] : 0.0f;
            __bf16 h0, l0, h1, l1;
            split_bf16(f0, h0, l0);
            split_bf16(f1, h1, l1);
            Ahi[c][2 * v]     = h0;  Alo[c][2 * v]     = l0;
            Ahi[c][2 * v + 1] = h1;  Alo[c][2 * v + 1] = l1;
        }
    }

    // ---- branchless store pointers: sentinel rows -> dummy scratch row ----
    // slot r covers row s_rows[r + 8*half], col n = wave*(Nn/NWAVES) + tile*16 + m
    float* pr[8];
    #pragma unroll
    for (int r = 0; r < 8; ++r) {
        int row = s_rows[r + 8 * half];
        float* base = (row >= 0) ? (out + (size_t)row * Nn) : dummy;
        pr[r] = base + (wave * (Nn / NWAVES) + m);
    }

    const v8u*   wpE = wp + (size_t)e * NT * 4 * 2 * 32;
    const float* be  = bias + (size_t)e * Nn;

    float runmax[8];
    #pragma unroll
    for (int r = 0; r < 8; ++r) runmax[r] = -3.402823466e38f;

    // ---- pass 1: WMMA logits from pre-packed W; bias folded into C init ----
    for (int tile = 0; tile < NT / NWAVES; ++tile) {
        const int ntile = wave * (NT / NWAVES) + tile;
        const int n     = ntile * 16 + m;
        const float bv  = be[n];
        v8f acc = {bv, bv, bv, bv, bv, bv, bv, bv};
        #pragma unroll
        for (int c = 0; c < 4; ++c) {
            size_t base = ((size_t)(ntile * 4 + c)) * 64;
            v16bf Bhi = __builtin_bit_cast(v16bf, wpE[base + lane]);
            v16bf Blo = __builtin_bit_cast(v16bf, wpE[base + 32 + lane]);
            acc = __builtin_amdgcn_wmma_f32_16x16x32_bf16(false, Ahi[c], false, Blo,
                                                          (short)0, acc, false, false);
            acc = __builtin_amdgcn_wmma_f32_16x16x32_bf16(false, Alo[c], false, Bhi,
                                                          (short)0, acc, false, false);
            acc = __builtin_amdgcn_wmma_f32_16x16x32_bf16(false, Ahi[c], false, Bhi,
                                                          (short)0, acc, false, false);
        }
        #pragma unroll
        for (int r = 0; r < 8; ++r) {   // C/D: slot r -> row r + 8*half, col n
            float vlog = acc[r];
            runmax[r] = fmaxf(runmax[r], vlog);
            *pr[r] = vlog;              // unconditional; sentinels hit dummy
            pr[r] += 16;                // next 16-col tile of this wave
        }
    }

    // ---- per-row max: reduce within 16-lane halves, then across waves ----
    #pragma unroll
    for (int r = 0; r < 8; ++r) {
        float v = runmax[r];
        v = fmaxf(v, __shfl_xor(v, 1, 32));
        v = fmaxf(v, __shfl_xor(v, 2, 32));
        v = fmaxf(v, __shfl_xor(v, 4, 32));
        v = fmaxf(v, __shfl_xor(v, 8, 32));
        runmax[r] = v;
    }
    if (m == 0) {
        #pragma unroll
        for (int r = 0; r < 8; ++r) s_pmax[wave][r + 8 * half] = runmax[r];
    }
    __threadfence();   // pass-1 logit stores visible before re-read
    __syncthreads();
    if (tid < 16) {
        float mm = s_pmax[0][tid];
        #pragma unroll
        for (int w = 1; w < NWAVES; ++w) mm = fmaxf(mm, s_pmax[w][tid]);
        s_rowmax[tid] = mm;
    }
    __syncthreads();

    // ---- pass 2a: sum of exp per row (32 threads/row, coalesced, L2 hits) ----
    {
        const int r   = tid >> 5;       // 16 rows, 32 threads each
        const int c0  = tid & 31;
        const int row = s_rows[r];
        const float M = s_rowmax[r];
        float s = 0.0f;
        if (row >= 0) {
            const float* orow = out + (size_t)row * Nn;
            for (int n = c0; n < Nn; n += 32) s += __expf(orow[n] - M);
        }
        s_psum[r][c0] = s;
    }
    __syncthreads();
    if (tid < 16) {
        float S = 0.0f;
        #pragma unroll
        for (int c = 0; c < 32; ++c) S += s_psum[tid][c];
        const int row = s_rows[tid];
        float vscale = (row >= 0) ? validf[row] : 0.0f;
        s_rowinv[tid] = (S > 0.0f) ? (vscale / S) : 0.0f;
    }
    __syncthreads();

    // ---- pass 2b: normalize in place; non-temporal final stores ----
    for (int r = 0; r < 16; ++r) {
        const int row = s_rows[r];
        if (row < 0) continue;
        const float M   = s_rowmax[r];
        const float inv = s_rowinv[r];
        float* orow = out + (size_t)row * Nn;
        for (int n = tid; n < Nn; n += NTHREADS) {
            float p = __expf(orow[n] - M) * inv;
            __builtin_nontemporal_store(p, &orow[n]);
        }
    }
}

extern "C" void kernel_launch(void* const* d_in, const int* in_sizes, int n_in,
                              void* d_out, int out_size, void* d_ws, size_t ws_size,
                              hipStream_t stream) {
    (void)in_sizes; (void)n_in; (void)out_size; (void)ws_size;
    const float* x   = (const float*)d_in[0];
    const float* W   = (const float*)d_in[1];
    const float* b   = (const float*)d_in[2];
    const float* tlo = (const float*)d_in[3];
    const float* thi = (const float*)d_in[4];
    float* out = (float*)d_out;

    char* wsb = (char*)d_ws;
    int*   counts = (int*)wsb;                                   // 8 ints (256B pad)
    int*   perm   = (int*)(wsb + 256);                           // E*B ints = 128KB
    float* validf = (float*)(wsb + 256 + Ecnt * Bsz * 4);        // B floats = 16KB
    float* dummy  = (float*)(wsb + 256 + Ecnt * Bsz * 4 + Bsz * 4);        // Nn floats = 32KB
    v8u*   wpack  = (v8u*)(wsb + 256 + Ecnt * Bsz * 4 + Bsz * 4 + Nn * 4); // 32MB, 32B-aligned

    init_counts_k<<<1, 32, 0, stream>>>(counts);
    route_k<<<(Bsz + 255) / 256, 256, 0, stream>>>(x, tlo, thi, counts, perm, validf);
    pack_w_k<<<Ecnt * NT, 128, 0, stream>>>(W, wpack);

    dim3 grid(1, Ecnt * TILES_PER_E, 1);
    moe_k<<<grid, NTHREADS, 0, stream>>>(x, wpack, b, counts, perm, validf, dummy, out);
}
